// GCN_Layer_sage_36120674959517
// MI455X (gfx1250) — matrix operand order repaired and checked by reading, not compiled, not run on hardware
//
#include <hip/hip_runtime.h>

#define DFEAT 128

typedef __attribute__((ext_vector_type(16))) _Float16 v16h;
typedef __attribute__((ext_vector_type(4)))  _Float16 v4h;
typedef __attribute__((ext_vector_type(8)))  float    v8f;

// ---------------------------------------------------------------------------
// fp32 -> (hi, lo) f16 split:  v ~= (float)hi + (float)lo,  ~2^-22 rel error.
// ---------------------------------------------------------------------------
__device__ __forceinline__ _Float16 split_hi(float v) {
    return (_Float16)v;
}
__device__ __forceinline__ _Float16 split_lo(float v) {
    _Float16 hi = (_Float16)v;
    return (_Float16)(v - (float)hi);
}

__device__ __forceinline__ void split4(float4 v, v4h& hi, v4h& lo) {
    v4h h, l;
    h[0] = split_hi(v.x); l[0] = split_lo(v.x);
    h[1] = split_hi(v.y); l[1] = split_lo(v.y);
    h[2] = split_hi(v.z); l[2] = split_lo(v.z);
    h[3] = split_hi(v.w); l[3] = split_lo(v.w);
    hi = h;
    lo = l;
}

__global__ void fill_zero_kernel(float* __restrict__ p, long long n) {
    long long i = (long long)blockIdx.x * blockDim.x + threadIdx.x;
    if (i < n) p[i] = 0.0f;
}

__global__ void degree_kernel(const int* __restrict__ dst, float* __restrict__ deg, int E) {
    int e = blockIdx.x * blockDim.x + threadIdx.x;
    if (e < E) unsafeAtomicAdd(&deg[dst[e]], 1.0f);   // native global_atomic_add_f32 at L2
}

__global__ void invdeg_kernel(float* __restrict__ deg, int n) {
    int i = blockIdx.x * blockDim.x + threadIdx.x;
    if (i < n) deg[i] = 1.0f / fmaxf(deg[i], 1.0f);
}

// Weights -> per-row planar hi/lo, natural k order (B-side layout):
// matrix row j occupies 256 halves: [j*256 + k] = hi, [j*256 + 128 + k] = lo.
__global__ void wconvert_kernel(const float* __restrict__ w, _Float16* __restrict__ out, int n) {
    int i = blockIdx.x * blockDim.x + threadIdx.x;
    if (i >= n) return;
    int j = i >> 7, k = i & 127;
    float v = w[i];
    out[(long long)j * 256 + k]       = split_hi(v);
    out[(long long)j * 256 + 128 + k] = split_lo(v);
}

// One wave (32 lanes) per edge; lane l handles features [4l, 4l+4).
// h table (51 MB) and msg (51 MB) are L2-resident (192 MB L2): gathers and
// f32 atomics run at L2 speed.  This is the roofline-dominant kernel.
__global__ void scatter_kernel(const float4* __restrict__ h4, const int* __restrict__ src,
                               const int* __restrict__ dst, float* __restrict__ msg, int E) {
    long long gid = (long long)blockIdx.x * blockDim.x + threadIdx.x;
    int e    = (int)(gid >> 5);
    int lane = (int)(gid & 31);
    if (e >= E) return;
    int s = src[e];
    int d = dst[e];
    float4 v = h4[(long long)s * (DFEAT / 4) + lane];
    float* out = msg + (long long)d * DFEAT + lane * 4;
    unsafeAtomicAdd(out + 0, v.x);
    unsafeAtomicAdd(out + 1, v.y);
    unsafeAtomicAdd(out + 2, v.z);
    unsafeAtomicAdd(out + 3, v.w);
}

// ---------------------------------------------------------------------------
// Row packer: one wave per node-row.
//  - h row  -> hpk   (A-side layout, separate buffer)
//  - msg row * invdeg -> mean, written IN PLACE over the same 512B row footprint
//    (hi halves at row*256+pos, lo at row*256+128+pos).  Safe: the whole fp32
//    row is read by ONE b128 wave-instruction whose results the stores depend
//    on, so all reads complete before any aliasing store issues, and no other
//    wave touches this row's bytes.
// A-side k order: natural, same as B side (fragment element s of lane half h is k = 16h + s per 32-k block)
// so each WMMA A fragment is one contiguous 32B v16h load.
// ---------------------------------------------------------------------------
__global__ void pack_rows_kernel(const float4* __restrict__ h4, float* msg,
                                 const float* __restrict__ invdeg,
                                 _Float16* __restrict__ hpk, int nrows) {
    long long gid = (long long)blockIdx.x * blockDim.x + threadIdx.x;
    int row  = (int)(gid >> 5);
    int lane = (int)(gid & 31);
    if (row >= nrows) return;
    const int k = lane * 4;                                              // 4 consecutive k
    const int p = k;                                                     // natural k order
    const float inv = invdeg[row];

    float4 hv = h4[(long long)row * (DFEAT / 4) + lane];
    float4 mv = ((const float4*)msg)[(long long)row * (DFEAT / 4) + lane];
    mv.x *= inv; mv.y *= inv; mv.z *= inv; mv.w *= inv;

    v4h hhi, hlo, mhi, mlo;
    split4(hv, hhi, hlo);
    split4(mv, mhi, mlo);

    _Float16* mrow = (_Float16*)msg + (long long)row * 256;
    _Float16* hrow = hpk            + (long long)row * 256;
    *(v4h*)(mrow + p)       = mhi;
    *(v4h*)(mrow + 128 + p) = mlo;
    *(v4h*)(hrow + p)       = hhi;
    *(v4h*)(hrow + 128 + p) = hlo;
}

#define WMMA_F16(A, B, C) \
    __builtin_amdgcn_wmma_f32_16x16x32_f16(false, (A), false, (B), (short)0, (C), false, false)

// out = relu?( mean @ Wl^T + h @ Wr^T + b ).  One wave per 16-row M-tile,
// full 128-wide N in 8 v8f accumulators.  Every fragment (A via per-row planar
// layout, B via per-j planar layout) is ONE aligned 32B v16h load -> the
// inner loop is pure vmem + v_wmma, no register shuffling.
// Split precision: acc += Ahi*Bhi + Alo*Bhi + Ahi*Blo  (fp32-class accuracy).
__global__ void __launch_bounds__(128)
sage_gemm_wmma_kernel(const _Float16* __restrict__ meanpack, const _Float16* __restrict__ hpack,
                      const _Float16* __restrict__ wl, const _Float16* __restrict__ wr,
                      const float* __restrict__ bias, float* __restrict__ out,
                      int mtiles, int do_relu) {
    const int wave  = threadIdx.x >> 5;
    const int lane  = threadIdx.x & 31;
    const int mtile = blockIdx.x * (blockDim.x >> 5) + wave;
    if (mtile >= mtiles) return;                 // wave-uniform: EXEC stays all-ones

    const int n0   = lane & 15;                  // N (and A-row) position within tile
    const int half = lane >> 4;                  // lane half selects K sub-block
    const long long rowA = (long long)(mtile * 16 + n0) * 256;

    const v8f vzero = {0.f, 0.f, 0.f, 0.f, 0.f, 0.f, 0.f, 0.f};
    v8f acc[8];
#pragma unroll
    for (int nt = 0; nt < 8; ++nt) acc[nt] = vzero;

    for (int kt = 0; kt < 4; ++kt) {
        const int fo = kt * 32 + half * 16;      // fragment offset within a 256-half row
        const v16h amh = *(const v16h*)(meanpack + rowA + fo);
        const v16h aml = *(const v16h*)(meanpack + rowA + fo + 128);
        const v16h ahh = *(const v16h*)(hpack    + rowA + fo);
        const v16h ahl = *(const v16h*)(hpack    + rowA + fo + 128);
#pragma unroll
        for (int nt = 0; nt < 8; ++nt) {
            const long long j = (long long)(nt * 16 + n0) * 256;   // W row = out feature
            const v16h blh = *(const v16h*)(wl + j + fo);
            const v16h bll = *(const v16h*)(wl + j + fo + 128);
            const v16h brh = *(const v16h*)(wr + j + fo);
            const v16h brl = *(const v16h*)(wr + j + fo + 128);
            acc[nt] = WMMA_F16(amh, blh, acc[nt]);   // mean @ Wl^T  hi*hi
            acc[nt] = WMMA_F16(aml, blh, acc[nt]);   //              lo*hi
            acc[nt] = WMMA_F16(amh, bll, acc[nt]);   //              hi*lo
            acc[nt] = WMMA_F16(ahh, brh, acc[nt]);   // h    @ Wr^T  hi*hi
            acc[nt] = WMMA_F16(ahl, brh, acc[nt]);   //              lo*hi
            acc[nt] = WMMA_F16(ahh, brl, acc[nt]);   //              hi*lo
        }
    }

    // C/D layout (32-bit 16x16): lane -> N = lane%16, VGPR i -> M = i + 8*(lane/16)
#pragma unroll
    for (int nt = 0; nt < 8; ++nt) {
        const float bv = bias[nt * 16 + n0];
#pragma unroll
        for (int i = 0; i < 8; ++i) {
            const int mrow = mtile * 16 + half * 8 + i;
            float v = acc[nt][i] + bv;
            if (do_relu) v = fmaxf(v, 0.0f);
            out[(long long)mrow * DFEAT + nt * 16 + n0] = v;
        }
    }
}

// ---------------------------------------------------------------------------
extern "C" void kernel_launch(void* const* d_in, const int* in_sizes, int n_in,
                              void* d_out, int out_size, void* d_ws, size_t ws_size,
                              hipStream_t stream) {
    const float* x  = (const float*)d_in[0];
    const int*   ei = (const int*)d_in[1];          // [2][E] (JAX x64 off -> int32)
    const int Nn = in_sizes[0] / DFEAT;             // 100000
    const int Ee = in_sizes[1] / 2;                 // 3200000
    const int* src = ei;
    const int* dst = ei + Ee;

    const long long nd = (long long)Nn * DFEAT;

    // Workspace (~155 MB): deg | msg fp32 (becomes packed mean in place) |
    // hpk (2*nd halves) | h1 fp32 | packed weights (6 * 32768 halves).
    float*     deg = (float*)d_ws;
    float*     msg = deg + Nn;
    _Float16*  hpk = (_Float16*)(msg + nd);
    float*     h1  = (float*)(hpk + 2 * nd);
    _Float16*  wpk = (_Float16*)(h1 + nd);
    const long long WMAT = (long long)DFEAT * 256;  // halves per packed matrix

    const int mtiles = (Nn + 15) / 16;              // 6250, no tail (100000 % 16 == 0)

    // Degrees: identical for all three layers, compute once.
    fill_zero_kernel<<<(Nn + 255) / 256, 256, 0, stream>>>(deg, Nn);
    degree_kernel<<<(Ee + 255) / 256, 256, 0, stream>>>(dst, deg, Ee);
    invdeg_kernel<<<(Nn + 255) / 256, 256, 0, stream>>>(deg, Nn);

    // Weights -> planar hi/lo (order: W1l W1r W2l W2r W3l W3r).
    for (int i = 0; i < 6; ++i)
        wconvert_kernel<<<(DFEAT * DFEAT + 255) / 256, 256, 0, stream>>>(
            (const float*)d_in[2 + i], wpk + i * WMAT, DFEAT * DFEAT);

    const float* hin[3]  = { x, h1, (float*)d_out };
    float*       hout[3] = { h1, (float*)d_out, (float*)d_out };

    for (int l = 0; l < 3; ++l) {
        fill_zero_kernel<<<(int)((nd + 255) / 256), 256, 0, stream>>>(msg, nd);
        scatter_kernel<<<(int)(((long long)Ee * 32 + 255) / 256), 256, 0, stream>>>(
            (const float4*)hin[l], src, dst, msg, Ee);
        pack_rows_kernel<<<(int)(((long long)Nn * 32 + 255) / 256), 256, 0, stream>>>(
            (const float4*)hin[l], msg, deg, hpk, Nn);
        sage_gemm_wmma_kernel<<<(mtiles + 3) / 4, 128, 0, stream>>>(
            (const _Float16*)msg, (const _Float16*)hpk,
            wpk + (long long)(2 * l) * WMAT, wpk + (long long)(2 * l + 1) * WMAT,
            (const float*)d_in[8 + l], hout[l], mtiles, (l < 2) ? 1 : 0);
    }
}
